// HierarchicalGraphMemoryNetwork_59493886984405
// MI455X (gfx1250) — matrix operand order repaired and checked by reading, not compiled, not run on hardware
//
#include <hip/hip_runtime.h>
#include <hip/hip_bf16.h>

// ---------------------------------------------------------------------------
// Problem constants (match reference)
// ---------------------------------------------------------------------------
#define TT   8192
#define INF_ 256
#define HH   256
#define RR   4
#define MEMN 256
#define MVD  128
#define OUTN 256
#define G4H  1024   // 4*H

typedef __attribute__((ext_vector_type(16))) _Float16 v16h;
typedef __attribute__((ext_vector_type(8)))  _Float16 v8h;
typedef __attribute__((ext_vector_type(8)))  float    v8f;

// ---------------------------------------------------------------------------
// f32 -> f16 cast
// ---------------------------------------------------------------------------
__global__ void cast_f32_to_f16(const float* __restrict__ src,
                                _Float16* __restrict__ dst, int n) {
    int i = blockIdx.x * blockDim.x + threadIdx.x;
    if (i < n) dst[i] = (_Float16)src[i];
}

// ---------------------------------------------------------------------------
// A-fragment load per CDNA5 ISA 7.12.2 (16x32 f16 A, wave32):
//   lane<16  : row=lane,    K = {0..7, 16..23}
//   lane>=16 : row=lane-16, K = {8..15, 24..31}
// p already points at (row, half*8); k is the K-tile base.
// ---------------------------------------------------------------------------
__device__ __forceinline__ v16h load_a_frag(const _Float16* __restrict__ p, int k) {
    v8h x0 = *(const v8h*)(p + k);        // K group {0..7} / {8..15}
    v8h x1 = *(const v8h*)(p + k + 16);   // K group {16..23} / {24..31}
    v16h v;
#pragma unroll
    for (int j = 0; j < 8; ++j) { v[j] = x0[j]; v[j + 8] = x1[j]; }
    return v;
}

#define WMMA_F16(A_, B_, C_) \
    __builtin_amdgcn_wmma_f32_16x16x32_f16(false, (A_), false, (B_), (short)0, (C_), false, false)

// ---------------------------------------------------------------------------
// WMMA GEMM:  C[M,N] (f32) = A[M,K] (f16, row-major) * B[N,K]^T (f16) + bias[N]
// Each wave computes a 32x32 C macro-tile (2 M-tiles x 2 N-tiles -> 4
// independent accumulators, 4 back-to-back v_wmma per 32-K step).
// Software pipeline: two structural fragment sets (A-set / B-set) ping-pong
// across a 64-wide K step -- every buffer is written once and read once per
// iteration (no rotation copies), K is always a multiple of 64 here, and the
// next half-step's loads are issued behind the current 4 WMMAs so the matrix
// pipe waits on a partial load count only.  Last-iteration "next" load is a
// branchless clamped re-read of tile 0 (valid, dead).
// B fragment (32x16 f16): lane<16 col=lane K={0..7,16..23}; lane>=16 col=lane-16
// K={8..15,24..31} (same as A).  C/D: VGPR v -> row v / v+8 by lane half.
// ---------------------------------------------------------------------------
__global__ void wmma_gemm_bias(const _Float16* __restrict__ A,
                               const _Float16* __restrict__ B,
                               const float* __restrict__ bias,
                               float* __restrict__ C,
                               int M, int N, int K) {
    const int lane = threadIdx.x & 31;
    const int wave = threadIdx.x >> 5;             // 0..3
    const int half = lane >> 4;                    // lane half
    const int r    = lane & 15;
    const int n0   = blockIdx.x * 128 + wave * 32; // wave's first column
    const int m0   = blockIdx.y * 32;              // block's first row

    const _Float16* Ap0 = A + (size_t)(m0 + r) * K + half * 8;
    const _Float16* Ap1 = Ap0 + (size_t)16 * K;
    const _Float16* Bp0 = B + (size_t)(n0 + r) * K + half * 8;
    const _Float16* Bp1 = Bp0 + (size_t)16 * K;

    v8f acc00 = {}, acc01 = {}, acc10 = {}, acc11 = {};

    // prologue: fragment set A at K tile 0
    v16h a0A = load_a_frag(Ap0, 0);
    v16h a1A = load_a_frag(Ap1, 0);
    v16h b0A = load_a_frag(Bp0, 0);
    v16h b1A = load_a_frag(Bp1, 0);

    const int K64 = K >> 6;                        // >= 4 for all call sites
    int k = 0;
    for (int it = 0; it < K64; ++it) {
        // fragment set B at k+32 (always in range)
        v16h a0B = load_a_frag(Ap0, k + 32);
        v16h a1B = load_a_frag(Ap1, k + 32);
        v16h b0B = load_a_frag(Bp0, k + 32);
        v16h b1B = load_a_frag(Bp1, k + 32);

        acc00 = WMMA_F16(a0A, b0A, acc00);
        acc01 = WMMA_F16(a0A, b1A, acc01);
        acc10 = WMMA_F16(a1A, b0A, acc10);
        acc11 = WMMA_F16(a1A, b1A, acc11);

        // fragment set A at k+64; branchless clamp on last iteration
        const int kn = (it + 1 < K64) ? (k + 64) : 0;
        __builtin_prefetch(Bp0 + kn + 32, 0, 1);   // -> global_prefetch_b8
        __builtin_prefetch(Ap0 + kn + 32, 0, 1);
        a0A = load_a_frag(Ap0, kn);
        a1A = load_a_frag(Ap1, kn);
        b0A = load_a_frag(Bp0, kn);
        b1A = load_a_frag(Bp1, kn);

        acc00 = WMMA_F16(a0B, b0B, acc00);
        acc01 = WMMA_F16(a0B, b1B, acc01);
        acc10 = WMMA_F16(a1B, b0B, acc10);
        acc11 = WMMA_F16(a1B, b1B, acc11);

        k += 64;
    }

    // epilogue: bias + store (C/D layout: VGPR v -> row v (+8 for lane half 1))
    const int   colA = n0 + r;
    const int   colB = n0 + 16 + r;
    const float bv0  = bias ? bias[colA] : 0.0f;
    const float bv1  = bias ? bias[colB] : 0.0f;
    float* C0 = C + (size_t)(m0 + half * 8) * N;       // M tile 0
    float* C1 = C + (size_t)(m0 + 16 + half * 8) * N;  // M tile 1
#pragma unroll
    for (int v = 0; v < 8; ++v) {
        C0[(size_t)v * N + colA] = acc00[v] + bv0;
        C0[(size_t)v * N + colB] = acc01[v] + bv1;
        C1[(size_t)v * N + colA] = acc10[v] + bv0;
        C1[(size_t)v * N + colB] = acc11[v] + bv1;
    }
}

// ---------------------------------------------------------------------------
// Persistent LSTM scan.  pre[T,1024] already holds Wih@x + (b_ih+b_hh).
// 1024 threads: thread j owns gate-row j (z_j = pre[t,j] + Whh[j,:]·h).
// h,c,z live in LDS; Whh (1 MB fp32) stays L2-resident.
// Gate order i,f,g,o (torch).  reverse!=0 walks t = T-1..0 (jax reverse scan).
// ---------------------------------------------------------------------------
__device__ __forceinline__ float sigmoidf_(float x) {
    return 1.0f / (1.0f + __expf(-x));
}

__global__ void lstm_scan_kernel(const float* __restrict__ pre,   // [T,1024]
                                 const float* __restrict__ Whh,   // [1024,256]
                                 float* __restrict__ hout,        // + col offset
                                 int hstride, int T, int reverse) {
    __shared__ float h[HH];
    __shared__ float c[HH];
    __shared__ float z[G4H];
    const int tid = threadIdx.x;                    // 0..1023
    if (tid < HH) { h[tid] = 0.0f; c[tid] = 0.0f; }
    __syncthreads();

    const float* w = Whh + (size_t)tid * HH;
    for (int s = 0; s < T; ++s) {
        const int t = reverse ? (T - 1 - s) : s;
        float acc = pre[(size_t)t * G4H + tid];
#pragma unroll 8
        for (int k = 0; k < HH; ++k) acc = fmaf(w[k], h[k], acc);
        z[tid] = acc;
        __syncthreads();
        if (tid < HH) {
            const float ig = sigmoidf_(z[tid]);
            const float fg = sigmoidf_(z[HH + tid]);
            const float gg = tanhf(z[2 * HH + tid]);
            const float og = sigmoidf_(z[3 * HH + tid]);
            const float cn = fg * c[tid] + ig * gg;
            const float hn = og * tanhf(cn);
            c[tid] = cn;
            h[tid] = hn;
            hout[(size_t)t * hstride + tid] = hn;
        }
        __syncthreads();
    }
}

// ---------------------------------------------------------------------------
// pair[t] = concat(nodes16[e0], nodes16[e1])   (grid=T, block=256)
// ---------------------------------------------------------------------------
__global__ void gather_pair_kernel(const _Float16* __restrict__ nodes16,
                                   const int* __restrict__ edges,
                                   _Float16* __restrict__ pair16) {
    const int t = blockIdx.x, j = threadIdx.x;
    const int e0 = edges[2 * t], e1 = edges[2 * t + 1];
    pair16[(size_t)t * 512 + j]       = nodes16[(size_t)e0 * HH + j];
    pair16[(size_t)t * 512 + 256 + j] = nodes16[(size_t)e1 * HH + j];
}

// relin[t] = concat(x16[t], f16(eh[t]))        (grid=T, block=256)
__global__ void build_relin_kernel(const _Float16* __restrict__ x16,
                                   const float* __restrict__ eh,
                                   _Float16* __restrict__ relin) {
    const int t = blockIdx.x, j = threadIdx.x;
    relin[(size_t)t * 512 + j]       = x16[(size_t)t * INF_ + j];
    relin[(size_t)t * 512 + 256 + j] = (_Float16)eh[(size_t)t * HH + j];
}

// ctrl[t] = concat(x16[t], f16(higher[t]), f16(memory[t]))  (grid=T, block=896)
__global__ void build_ctrl_kernel(const _Float16* __restrict__ x16,
                                  const float* __restrict__ higher,
                                  const float* __restrict__ mem,
                                  _Float16* __restrict__ ctrl) {
    const int t = blockIdx.x, j = threadIdx.x;
    _Float16 v;
    if (j < 256)      v = x16[(size_t)t * INF_ + j];
    else if (j < 768) v = (_Float16)higher[(size_t)t * 512 + (j - 256)];
    else              v = (_Float16)mem[(size_t)t * MVD + (j - 768)];
    ctrl[(size_t)t * 896 + j] = v;
}

// ---------------------------------------------------------------------------
// Host orchestration
// ---------------------------------------------------------------------------
static inline void launch_cast(const float* s, _Float16* d, int n, hipStream_t st) {
    cast_f32_to_f16<<<(n + 255) / 256, 256, 0, st>>>(s, d, n);
}
static inline void launch_gemm(float* C, const _Float16* A, const _Float16* B,
                               const float* bias, int M, int N, int K,
                               hipStream_t st) {
    dim3 grid(N / 128, M / 32);
    wmma_gemm_bias<<<grid, 128, 0, st>>>(A, B, bias, C, M, N, K);
}

extern "C" void kernel_launch(void* const* d_in, const int* in_sizes, int n_in,
                              void* d_out, int out_size, void* d_ws, size_t ws_size,
                              hipStream_t stream) {
    const float* inputs = (const float*)d_in[0];
    const int*   edges  = (const int*)d_in[1];
    const float* memory = (const float*)d_in[2];
    const float* Wl_ih  = (const float*)d_in[3];
    const float* Wl_hh  = (const float*)d_in[4];
    const float* bl     = (const float*)d_in[5];
    const float* Wh_ih  = (const float*)d_in[6];
    const float* Wh_hh  = (const float*)d_in[7];
    const float* bh     = (const float*)d_in[8];
    const float* Win    = (const float*)d_in[9];
    const float* bin_   = (const float*)d_in[10];
    const float* We     = (const float*)d_in[11];
    const float* be     = (const float*)d_in[12];
    const float* Wn_ih  = (const float*)d_in[13];
    const float* Wn_hh  = (const float*)d_in[14];
    const float* bn     = (const float*)d_in[15];
    const float* Wmem   = (const float*)d_in[16];
    const float* bmem   = (const float*)d_in[17];
    const float* Wc_ih  = (const float*)d_in[18];
    const float* Wc_hh  = (const float*)d_in[19];
    const float* bc     = (const float*)d_in[20];
    const float* Wout   = (const float*)d_in[21];
    const float* bout   = (const float*)d_in[22];

    float* out  = (float*)d_out;                    // [T, OUT]
    float* memv = out + (size_t)TT * OUTN;          // [T, MEM]

    // bump allocator over d_ws (deterministic, recomputed every call)
    char*  base = (char*)d_ws;
    size_t off  = 0;
    auto alloc = [&](size_t bytes) -> void* {
        void* p = base + off;
        off = (off + bytes + 255) & ~(size_t)255;
        return p;
    };
    _Float16* x16     = (_Float16*)alloc((size_t)TT * INF_ * 2);
    _Float16* lower16 = (_Float16*)alloc((size_t)TT * 512 * 2);
    _Float16* nodes16 = (_Float16*)alloc((size_t)TT * HH * 2);
    _Float16* pair16  = (_Float16*)alloc((size_t)TT * 512 * 2);
    _Float16* relin16 = (_Float16*)alloc((size_t)TT * 512 * 2);
    _Float16* ctrl16  = (_Float16*)alloc((size_t)TT * 896 * 2);
    _Float16* ch16    = (_Float16*)alloc((size_t)TT * HH * 2);
    _Float16* wl16    = (_Float16*)alloc((size_t)2 * G4H * INF_ * 2);
    _Float16* wh16    = (_Float16*)alloc((size_t)2 * G4H * 512 * 2);
    _Float16* win16   = (_Float16*)alloc((size_t)HH * INF_ * 2);
    _Float16* we16    = (_Float16*)alloc((size_t)RR * HH * 512 * 2);
    _Float16* wn16    = (_Float16*)alloc((size_t)RR * G4H * 512 * 2);
    _Float16* wmem16  = (_Float16*)alloc((size_t)MEMN * HH * 2);
    _Float16* wc16    = (_Float16*)alloc((size_t)G4H * 896 * 2);
    _Float16* wout16  = (_Float16*)alloc((size_t)OUTN * HH * 2);
    float* preF    = (float*)alloc((size_t)TT * G4H * 4);
    float* lowerF  = (float*)alloc((size_t)TT * 512 * 4);
    float* higherF = (float*)alloc((size_t)TT * 512 * 4);
    float* nodesF  = (float*)alloc((size_t)TT * HH * 4);
    float* ehF     = (float*)alloc((size_t)TT * HH * 4);
    float* ctrlhF  = (float*)alloc((size_t)TT * HH * 4);
    (void)ws_size; (void)in_sizes; (void)n_in; (void)out_size;

    // ---- one-time-per-call f16 conversions --------------------------------
    launch_cast(inputs, x16, TT * INF_, stream);
    launch_cast(Wl_ih, wl16, 2 * G4H * INF_, stream);
    launch_cast(Wh_ih, wh16, 2 * G4H * 512, stream);
    launch_cast(Win, win16, HH * INF_, stream);
    launch_cast(We, we16, RR * HH * 512, stream);
    launch_cast(Wn_ih, wn16, RR * G4H * 512, stream);
    launch_cast(Wmem, wmem16, MEMN * HH, stream);
    launch_cast(Wc_ih, wc16, G4H * 896, stream);
    launch_cast(Wout, wout16, OUTN * HH, stream);

    // ---- lower BiLSTM -----------------------------------------------------
    for (int dir = 0; dir < 2; ++dir) {
        launch_gemm(preF, x16, wl16 + (size_t)dir * G4H * INF_,
                    bl + dir * G4H, TT, G4H, INF_, stream);
        lstm_scan_kernel<<<1, G4H, 0, stream>>>(
            preF, Wl_hh + (size_t)dir * G4H * HH,
            lowerF + dir * HH, 512, TT, dir);
    }
    launch_cast(lowerF, lower16, TT * 512, stream);

    // ---- higher BiLSTM ----------------------------------------------------
    for (int dir = 0; dir < 2; ++dir) {
        launch_gemm(preF, lower16, wh16 + (size_t)dir * G4H * 512,
                    bh + dir * G4H, TT, G4H, 512, stream);
        lstm_scan_kernel<<<1, G4H, 0, stream>>>(
            preF, Wh_hh + (size_t)dir * G4H * HH,
            higherF + dir * HH, 512, TT, dir);
    }

    // ---- node init + relation rounds --------------------------------------
    launch_gemm(nodesF, x16, win16, bin_, TT, HH, INF_, stream);
    for (int r = 0; r < RR; ++r) {
        launch_cast(nodesF, nodes16, TT * HH, stream);
        gather_pair_kernel<<<TT, 256, 0, stream>>>(nodes16, edges, pair16);
        launch_gemm(ehF, pair16, we16 + (size_t)r * HH * 512,
                    be + r * HH, TT, HH, 512, stream);
        build_relin_kernel<<<TT, 256, 0, stream>>>(x16, ehF, relin16);
        launch_gemm(preF, relin16, wn16 + (size_t)r * G4H * 512,
                    bn + r * G4H, TT, G4H, 512, stream);
        lstm_scan_kernel<<<1, G4H, 0, stream>>>(
            preF, Wn_hh + (size_t)r * G4H * HH, nodesF, HH, TT, 0);
    }

    // ---- memory projection (second output) --------------------------------
    launch_cast(nodesF, nodes16, TT * HH, stream);
    launch_gemm(memv, nodes16, wmem16, bmem, TT, MEMN, HH, stream);

    // ---- control LSTM + output projection (first output) ------------------
    build_ctrl_kernel<<<TT, 896, 0, stream>>>(x16, higherF, memory, ctrl16);
    launch_gemm(preF, ctrl16, wc16, bc, TT, G4H, 896, stream);
    lstm_scan_kernel<<<1, G4H, 0, stream>>>(preF, Wc_hh, ctrlhF, HH, TT, 0);
    launch_cast(ctrlhF, ch16, TT * HH, stream);
    launch_gemm(out, ch16, wout16, bout, TT, OUTN, HH, stream);
}
